// GNN_53893249630287
// MI455X (gfx1250) — matrix operand order repaired and checked by reading, not compiled, not run on hardware
//
#include <hip/hip_runtime.h>
#include <hip/hip_bf16.h>

#define N_NODES 50000
#define N_EDGES 800000
#define N_GRAPHS 256
#define IN_DIM 128
#define HID 256
#define OUTD 32
#define BN_EPS 1e-5f

typedef __bf16 bf16;
typedef __attribute__((ext_vector_type(8)))  bf16  v8bf;
typedef __attribute__((ext_vector_type(16))) bf16  v16bf;
typedef __attribute__((ext_vector_type(8)))  float v8f;

__device__ __forceinline__ bf16 f2bf(float f) { return (bf16)f; }

__device__ __forceinline__ v8bf pack8(float4 a, float4 b) {
    v8bf r;
    r[0] = (bf16)a.x; r[1] = (bf16)a.y; r[2] = (bf16)a.z; r[3] = (bf16)a.w;
    r[4] = (bf16)b.x; r[5] = (bf16)b.y; r[6] = (bf16)b.z; r[7] = (bf16)b.w;
    return r;
}

// ---------------------------------------------------------------- utilities
__global__ void zero_kernel(float* __restrict__ p, size_t n) {
    size_t i = (size_t)blockIdx.x * blockDim.x + threadIdx.x;
    if (i < n) p[i] = 0.0f;
}

__global__ void deg_kernel(const int* __restrict__ dst, float* __restrict__ deg) {
    int e = blockIdx.x * blockDim.x + threadIdx.x;
    if (e < N_EDGES) atomicAdd(&deg[dst[e]], 1.0f);
}

__global__ void invdeg_kernel(float* __restrict__ deg) {
    int i = blockIdx.x * blockDim.x + threadIdx.x;
    if (i < N_NODES) {
        float d = deg[i];
        deg[i] = (d > 0.0f) ? (1.0f / d) : 0.0f;
    }
}

// scatter-add feat[src[e]] into hsum[dst[e]]  (float4 per thread, f32 atomics)
__global__ void aggregate_kernel(const float* __restrict__ feat,
                                 const int* __restrict__ src,
                                 const int* __restrict__ dst,
                                 float* __restrict__ hsum, int D, int D4) {
    size_t idx = (size_t)blockIdx.x * blockDim.x + threadIdx.x;
    size_t total = (size_t)N_EDGES * (size_t)D4;
    if (idx >= total) return;
    int e = (int)(idx / (size_t)D4);
    int c = (int)(idx % (size_t)D4);
    int s = src[e], d = dst[e];
    float4 v = ((const float4*)(feat + (size_t)s * D))[c];
    float* o = hsum + (size_t)d * D + (size_t)c * 4;
    atomicAdd(o + 0, v.x);
    atomicAdd(o + 1, v.y);
    atomicAdd(o + 2, v.z);
    atomicAdd(o + 3, v.w);
}

// h[row,:] *= inv_deg[row]
__global__ void scale_kernel(float* __restrict__ h, const float* __restrict__ inv_deg,
                             int D, int D4) {
    size_t idx = (size_t)blockIdx.x * blockDim.x + threadIdx.x;
    size_t total = (size_t)N_NODES * (size_t)D4;
    if (idx >= total) return;
    int row = (int)(idx / (size_t)D4);
    int c   = (int)(idx % (size_t)D4);
    float s = inv_deg[row];
    float4* p = ((float4*)(h + (size_t)row * D)) + c;
    float4 v = *p;
    v.x *= s; v.y *= s; v.z *= s; v.w *= s;
    *p = v;
}

// ---------------------------------------------------------------- fused SAGE GEMM
// D[M,N] = BN(ReLU(A0@W0 + A1@W1 + bias)); optionally store H; atomic pool add.
// Block tile 128x128, 256 threads = 8 waves (wave32), wave grid 4(M) x 2(N),
// each wave: 2x4 tiles of v_wmma_f32_16x16x32_bf16.
#define BM 128
#define BN 128
#define KT 32
#define LDA_S 40   // halfs; 80B row pitch -> 16B alignment for all b128 LDS ops
#define LDB_S 40

__global__ __launch_bounds__(256)
void sage_gemm_kernel(const float* __restrict__ A0, const float* __restrict__ A1,
                      const float* __restrict__ W0, const float* __restrict__ W1,
                      const float* __restrict__ bias,
                      const float* __restrict__ bn_g, const float* __restrict__ bn_b,
                      const float* __restrict__ bn_m, const float* __restrict__ bn_v,
                      const int* __restrict__ graph_ids,
                      float* __restrict__ H,      // nullptr -> skip store
                      float* __restrict__ pool,   // [N_GRAPHS x N] atomic accumulate
                      int M, int N, int K)        // K per A source (total 2K)
{
    __shared__ bf16 lds_a[BM * LDA_S];
    __shared__ bf16 lds_bt[BN * LDB_S];   // B stored transposed: [col][k]

    const int tid    = threadIdx.x;
    const int lane   = tid & 31;
    const int waveId = tid >> 5;
    const int wm     = waveId >> 1;   // 0..3
    const int wn     = waveId & 1;    // 0..1
    const int hi     = lane >> 4;     // 0/1
    const int l16    = lane & 15;

    const int rowBase = blockIdx.x * BM;
    const int colBase = blockIdx.y * BN;

    v8f acc[2][4];
#pragma unroll
    for (int rt = 0; rt < 2; ++rt)
#pragma unroll
        for (int ct = 0; ct < 4; ++ct)
#pragma unroll
            for (int v = 0; v < 8; ++v) acc[rt][ct][v] = 0.0f;

    // staging assignments (constant across k-loop)
    const int ar  = tid >> 1;             // A: row 0..127
    const int akc = (tid & 1) * 16;       // A: k-chunk 0 or 16
    const int bc  = tid & 127;            // B: column 0..127
    const int bkh = (tid >> 7) * 16;      // B: k-half 0 or 16

    for (int half = 0; half < 2; ++half) {
        const float* __restrict__ A = half ? A1 : A0;
        const float* __restrict__ W = half ? W1 : W0;

        for (int kb = 0; kb < K; kb += KT) {
            // --- stage A tile [BM x KT] fp32 -> bf16, two b128 stores/thread ---
            {
                int grow = rowBase + ar;
                v8bf* dstp = (v8bf*)(lds_a + ar * LDA_S + akc);
                if (grow < M) {
                    const float* srcp = A + (size_t)grow * K + kb + akc;
                    if (kb + KT < K) __builtin_prefetch(srcp + KT, 0, 0);
                    float4 f0 = *(const float4*)(srcp + 0);
                    float4 f1 = *(const float4*)(srcp + 4);
                    float4 f2 = *(const float4*)(srcp + 8);
                    float4 f3 = *(const float4*)(srcp + 12);
                    dstp[0] = pack8(f0, f1);
                    dstp[1] = pack8(f2, f3);
                } else {
                    v8bf z = {};
                    dstp[0] = z;
                    dstp[1] = z;
                }
            }
            // --- stage B tile [KT x BN] -> bf16 transposed, two b128 stores ---
            {
                // thread owns one column, 16 contiguous k values
                const float* srcp = W + (size_t)(kb + bkh) * N + colBase + bc;
                float4 w0, w1, w2, w3;
#pragma unroll
                for (int j = 0; j < 4; ++j) {
                    (&w0.x)[j] = srcp[(size_t)(j + 0) * N];
                    (&w1.x)[j] = srcp[(size_t)(j + 4) * N];
                    (&w2.x)[j] = srcp[(size_t)(j + 8) * N];
                    (&w3.x)[j] = srcp[(size_t)(j + 12) * N];
                }
                v8bf* dstp = (v8bf*)(lds_bt + bc * LDB_S + bkh);
                dstp[0] = pack8(w0, w1);
                dstp[1] = pack8(w2, w3);
            }
            __syncthreads();

            // --- fragments in contiguous k order (k = hi*16 + s) for both A and B ---
            v16bf fa[2], fb[4];
#pragma unroll
            for (int rt = 0; rt < 2; ++rt) {
                const bf16* p = lds_a + (wm * 32 + rt * 16 + l16) * LDA_S + hi * 16;
                v8bf c0 = *(const v8bf*)p;          // K = hi*16 .. +7
                v8bf c1 = *(const v8bf*)(p + 8);    // K = hi*16+8 .. +15
                fa[rt] = __builtin_shufflevector(c0, c1,
                         0,1,2,3,4,5,6,7,8,9,10,11,12,13,14,15);
            }
#pragma unroll
            for (int ct = 0; ct < 4; ++ct) {
                const bf16* p = lds_bt + (wn * 64 + ct * 16 + l16) * LDB_S + hi * 16;
                v8bf c0 = *(const v8bf*)p;          // K = hi*16 .. +7
                v8bf c1 = *(const v8bf*)(p + 8);    // K = hi*16+8 .. +15
                fb[ct] = __builtin_shufflevector(c0, c1,
                         0,1,2,3,4,5,6,7,8,9,10,11,12,13,14,15);
            }
#pragma unroll
            for (int rt = 0; rt < 2; ++rt)
#pragma unroll
                for (int ct = 0; ct < 4; ++ct)
                    acc[rt][ct] = __builtin_amdgcn_wmma_f32_16x16x32_bf16(
                        false, fa[rt], false, fb[ct],
                        (short)0, acc[rt][ct], false, false);
            __syncthreads();
        }
    }

    // --- epilogue: bias + ReLU + BN(eval) + optional store + pooled atomicAdd ---
#pragma unroll
    for (int ct = 0; ct < 4; ++ct) {
        int gcol = colBase + wn * 64 + ct * 16 + l16;
        float sc = bn_g[gcol] * rsqrtf(bn_v[gcol] + BN_EPS);
        float mm = bn_m[gcol];
        float bb = bn_b[gcol];
        float bi = bias[gcol];
#pragma unroll
        for (int rt = 0; rt < 2; ++rt) {
#pragma unroll
            for (int v = 0; v < 8; ++v) {
                int grow = rowBase + wm * 32 + rt * 16 + hi * 8 + v;
                if (grow < M) {
                    float x = acc[rt][ct][v] + bi;
                    x = fmaxf(x, 0.0f);
                    x = (x - mm) * sc + bb;
                    if (H) H[(size_t)grow * N + gcol] = x;
                    atomicAdd(&pool[graph_ids[grow] * N + gcol], x);
                }
            }
        }
    }
}

// ---------------------------------------------------------------- readout
// one wave per graph; lane = output class. Then log_softmax across 32 lanes.
__global__ __launch_bounds__(32)
void readout_kernel(const float* __restrict__ p1, const float* __restrict__ p2,
                    const float* __restrict__ Wp1, const float* __restrict__ bp1,
                    const float* __restrict__ g1, const float* __restrict__ b1,
                    const float* __restrict__ m1, const float* __restrict__ v1,
                    const float* __restrict__ Wp2, const float* __restrict__ bp2,
                    const float* __restrict__ g2, const float* __restrict__ b2,
                    const float* __restrict__ m2, const float* __restrict__ v2,
                    float* __restrict__ out)
{
    int g = blockIdx.x;
    int lane = threadIdx.x;   // 0..31
    const float* r1 = p1 + (size_t)g * HID;
    const float* r2 = p2 + (size_t)g * HID;
    float a1 = bp1[lane], a2 = bp2[lane];
    for (int k = 0; k < HID; ++k) {
        float s1 = r1[k], s2 = r2[k];
        a1 = fmaf(s1, Wp1[k * OUTD + lane], a1);
        a2 = fmaf(s2, Wp2[k * OUTD + lane], a2);
    }
    a1 = fmaxf(a1, 0.0f);
    a2 = fmaxf(a2, 0.0f);
    float o1 = (a1 - m1[lane]) * (g1[lane] * rsqrtf(v1[lane] + BN_EPS)) + b1[lane];
    float o2 = (a2 - m2[lane]) * (g2[lane] * rsqrtf(v2[lane] + BN_EPS)) + b2[lane];
    float o = 0.3f * o1 + 0.6f * o2;

    float mx = o;
#pragma unroll
    for (int off = 16; off > 0; off >>= 1) mx = fmaxf(mx, __shfl_xor(mx, off, 32));
    float ex = expf(o - mx);
    float s = ex;
#pragma unroll
    for (int off = 16; off > 0; off >>= 1) s += __shfl_xor(s, off, 32);
    out[(size_t)g * OUTD + lane] = (o - mx) - logf(s);
}

// ---------------------------------------------------------------- launch
extern "C" void kernel_launch(void* const* d_in, const int* in_sizes, int n_in,
                              void* d_out, int out_size, void* d_ws, size_t ws_size,
                              hipStream_t stream) {
    const float* feat  = (const float*)d_in[0];
    const int*   src   = (const int*)d_in[1];
    const int*   dst   = (const int*)d_in[2];
    const int*   gids  = (const int*)d_in[3];
    const float* Ws1   = (const float*)d_in[4];
    const float* Wn1   = (const float*)d_in[5];
    const float* b1    = (const float*)d_in[6];
    const float* bn1g  = (const float*)d_in[7];
    const float* bn1b  = (const float*)d_in[8];
    const float* bn1m  = (const float*)d_in[9];
    const float* bn1v  = (const float*)d_in[10];
    const float* Ws2   = (const float*)d_in[11];
    const float* Wn2   = (const float*)d_in[12];
    const float* b2    = (const float*)d_in[13];
    const float* bn2g  = (const float*)d_in[14];
    const float* bn2b  = (const float*)d_in[15];
    const float* bn2m  = (const float*)d_in[16];
    const float* bn2v  = (const float*)d_in[17];
    const float* Wp1   = (const float*)d_in[18];
    const float* bp1   = (const float*)d_in[19];
    const float* pg1   = (const float*)d_in[20];
    const float* pb1   = (const float*)d_in[21];
    const float* pm1   = (const float*)d_in[22];
    const float* pv1   = (const float*)d_in[23];
    const float* Wp2   = (const float*)d_in[24];
    const float* bp2   = (const float*)d_in[25];
    const float* pg2   = (const float*)d_in[26];
    const float* pb2   = (const float*)d_in[27];
    const float* pm2   = (const float*)d_in[28];
    const float* pv2   = (const float*)d_in[29];
    float* out = (float*)d_out;

    // workspace layout (floats)
    float* ws      = (float*)d_ws;
    float* inv_deg = ws;                               // 50000 (padded to 50048)
    float* hn1     = inv_deg + 50048;                  // 50000*128
    float* h1      = hn1 + (size_t)N_NODES * IN_DIM;   // 50000*256
    float* hn2     = h1  + (size_t)N_NODES * HID;      // 50000*256
    float* p1      = hn2 + (size_t)N_NODES * HID;      // 256*256
    float* p2      = p1  + N_GRAPHS * HID;             // 256*256

    const int TB = 256;
    auto blocks = [](size_t n, int tb) { return (unsigned)((n + tb - 1) / tb); };

    // zero accumulators (every call: graph replays must be deterministic)
    zero_kernel<<<blocks(N_NODES, TB), TB, 0, stream>>>(inv_deg, N_NODES);
    zero_kernel<<<blocks((size_t)N_NODES * IN_DIM, TB), TB, 0, stream>>>(
        hn1, (size_t)N_NODES * IN_DIM);
    zero_kernel<<<blocks((size_t)N_NODES * HID, TB), TB, 0, stream>>>(
        hn2, (size_t)N_NODES * HID);
    zero_kernel<<<blocks((size_t)2 * N_GRAPHS * HID, TB), TB, 0, stream>>>(
        p1, (size_t)2 * N_GRAPHS * HID);  // p1 and p2 are contiguous

    // degrees
    deg_kernel<<<blocks(N_EDGES, TB), TB, 0, stream>>>(dst, inv_deg);
    invdeg_kernel<<<blocks(N_NODES, TB), TB, 0, stream>>>(inv_deg);

    // layer 1 aggregation: hn1 = mean_{j->i} feat[j]
    aggregate_kernel<<<blocks((size_t)N_EDGES * (IN_DIM / 4), TB), TB, 0, stream>>>(
        feat, src, dst, hn1, IN_DIM, IN_DIM / 4);
    scale_kernel<<<blocks((size_t)N_NODES * (IN_DIM / 4), TB), TB, 0, stream>>>(
        hn1, inv_deg, IN_DIM, IN_DIM / 4);

    // layer 1 GEMM: h1 = BN(ReLU(feat@Ws1 + hn1@Wn1 + b1)); pool into p1
    {
        dim3 grid((N_NODES + BM - 1) / BM, HID / BN);
        sage_gemm_kernel<<<grid, 256, 0, stream>>>(
            feat, hn1, Ws1, Wn1, b1, bn1g, bn1b, bn1m, bn1v,
            gids, h1, p1, N_NODES, HID, IN_DIM);
    }

    // layer 2 aggregation: hn2 = mean_{j->i} h1[j]
    aggregate_kernel<<<blocks((size_t)N_EDGES * (HID / 4), TB), TB, 0, stream>>>(
        h1, src, dst, hn2, HID, HID / 4);
    scale_kernel<<<blocks((size_t)N_NODES * (HID / 4), TB), TB, 0, stream>>>(
        hn2, inv_deg, HID, HID / 4);

    // layer 2 GEMM: h2 only feeds p2 -> pool directly, never store h2
    {
        dim3 grid((N_NODES + BM - 1) / BM, HID / BN);
        sage_gemm_kernel<<<grid, 256, 0, stream>>>(
            h1, hn2, Ws2, Wn2, b2, bn2g, bn2b, bn2m, bn2v,
            gids, nullptr, p2, N_NODES, HID, HID);
    }

    // readout + log_softmax
    readout_kernel<<<N_GRAPHS, 32, 0, stream>>>(
        p1, p2, Wp1, bp1, pg1, pb1, pm1, pv1,
        Wp2, bp2, pg2, pb2, pm2, pv2, out);
}
